// MultiHeadAttention_8452495638766
// MI455X (gfx1250) — matrix operand order repaired and checked by reading, not compiled, not run on hardware
//
#include <hip/hip_runtime.h>

typedef __attribute__((ext_vector_type(16))) __bf16 v16bf;
typedef __attribute__((ext_vector_type(8)))  __bf16 v8bf;
typedef __attribute__((ext_vector_type(8)))  float  v8f;

namespace {
constexpr int BB = 2, SS = 2048, EE = 2048, HH = 16, GG = 4, DD = 128;
constexpr int EPAD = EE + 8;  // +8 bf16 (16B) row pad: lane m -> LDS bank 4m
constexpr float SCALE = 0.08838834764831845f;  // 1/sqrt(D)
constexpr float NEG   = -3.0e38f;
}

// ---------------- WMMA fragment helpers (CDNA5 ISA 7.12.2, wave32) ----------
// A 16x32 bf16: lane = row M (mod 16); lanes<16 hold K {0..7,16..23},
// lanes>=16 hold K {8..15,24..31}, ascending within the 16 halves.
static __device__ inline v16bf load_a_bf16(const __bf16* __restrict__ row, int half) {
  v8bf lo = *(const v8bf*)(row + half * 8);
  v8bf hi = *(const v8bf*)(row + half * 8 + 16);
  v16bf a;
#pragma unroll
  for (int i = 0; i < 8; ++i) { a[i] = lo[i]; a[8 + i] = hi[i]; }
  return a;
}

// B 32x16 bf16: lane = column N (mod 16); lanes<16 hold K {0..7,16..23},
// lanes>=16 hold K {8..15,24..31}. `col` points at this lane's column, unit K stride.
static __device__ inline v16bf load_b_bf16(const __bf16* __restrict__ col, int half) {
  v8bf lo = *(const v8bf*)(col + half * 8);
  v8bf hi = *(const v8bf*)(col + half * 8 + 16);
  v16bf b;
#pragma unroll
  for (int i = 0; i < 8; ++i) { b[i] = lo[i]; b[8 + i] = hi[i]; }
  return b;
}

static __device__ inline v8f wmma_bf16(v16bf a, v16bf b, v8f c) {
  return __builtin_amdgcn_wmma_f32_16x16x32_bf16(false, a, false, b, (short)0, c,
                                                 false, false);
}

// Async global->LDS copy of 16 bytes (CDNA5, tracked by ASYNCcnt).
static __device__ inline void async_copy_b128(unsigned lds_byte_off,
                                              const void* gaddr) {
  asm volatile("global_load_async_to_lds_b128 %0, %1, off"
               :
               : "v"(lds_byte_off), "v"(gaddr)
               : "memory");
}
static __device__ inline void wait_asynccnt0() {
  asm volatile("s_wait_asynccnt 0" ::: "memory");
}

// ---------------- kernel 0: x f32 -> bf16 (one-time, removes 24x re-read
// amplification and the per-WMMA cvt chain in the projection loop) ----------
__global__ void k_xconv(const float* __restrict__ x, __bf16* __restrict__ xb) {
  const size_t i = ((size_t)blockIdx.x * 256 + threadIdx.x) * 8;
  float4 f0 = *(const float4*)(x + i);
  float4 f1 = *(const float4*)(x + i + 4);
  v8bf o;
  o[0] = (__bf16)f0.x; o[1] = (__bf16)f0.y; o[2] = (__bf16)f0.z; o[3] = (__bf16)f0.w;
  o[4] = (__bf16)f1.x; o[5] = (__bf16)f1.y; o[6] = (__bf16)f1.z; o[7] = (__bf16)f1.w;
  *(v8bf*)(xb + i) = o;
}

// ---------------- kernel 1: weight transpose + bf16 convert ------------------
// in: [nh, E, D] f32  ->  out: [nh, D, E] bf16   (coalesced both directions)
__global__ void k_wtrans(const float* __restrict__ in, __bf16* __restrict__ out) {
  __shared__ float tile[32][33];
  const int tx = threadIdx.x, ty = threadIdx.y;
  const int d0 = blockIdx.x * 32, e0 = blockIdx.y * 32, h = blockIdx.z;
  const float* src = in + (size_t)h * EE * DD;
  __bf16* dst = out + (size_t)h * DD * EE;
#pragma unroll
  for (int i = 0; i < 4; ++i)
    tile[ty + 8 * i][tx] = src[(size_t)(e0 + ty + 8 * i) * DD + d0 + tx];
  __syncthreads();
#pragma unroll
  for (int i = 0; i < 4; ++i)
    dst[(size_t)(d0 + ty + 8 * i) * EE + e0 + tx] = (__bf16)tile[tx][ty + 8 * i];
}

// ---------------- kernel 2: fused QKV projection (bf16 WMMA, f32 accum) ------
// grid: (S/16, H + 2G, B), block 256 = 8 waves; wave w owns d-slice [16w,16w+16).
// The shared 16-token A-tile (16 x 2048 bf16 = 64KB of the 320KB WGP LDS) is
// staged ONCE per block via async global->LDS, then all 8 waves stream their
// K-loop A-fragments from LDS instead of 8x-redundant global loads.
__global__ void __launch_bounds__(256) k_qkv(
    const __bf16* __restrict__ xb, const __bf16* __restrict__ WqT,
    const __bf16* __restrict__ WkT, const __bf16* __restrict__ WvT,
    const float* __restrict__ bq, const float* __restrict__ bk,
    const float* __restrict__ bv, __bf16* __restrict__ Qb,
    __bf16* __restrict__ Kb, __bf16* __restrict__ Vt) {
  __shared__ __align__(16) __bf16 xtile[16][EPAD];
  const int lane = threadIdx.x & 31, wave = threadIdx.x >> 5;
  const int n = lane & 15, half = lane >> 4;
  const int s0 = blockIdx.x * 16, vh = blockIdx.y, b = blockIdx.z;
  const int n0 = wave * 16;

  // ---- async stage of the x tile: 4096 chunks of 16B, 16 per thread ----
  {
    const __bf16* gbase = xb + ((size_t)b * SS + s0) * EE;
#pragma unroll
    for (int i = 0; i < 16; ++i) {
      const int c = threadIdx.x + 256 * i;   // chunk id
      const int row = c >> 8;                // 256 chunks (4096B) per row
      const int col8 = (c & 255) * 8;        // element offset within row
      const unsigned lds_off =
          (unsigned)(uintptr_t)(&xtile[row][col8]);  // low 32b == LDS address
      async_copy_b128(lds_off, gbase + (size_t)row * EE + col8);
    }
    wait_asynccnt0();
    __syncthreads();
  }

  const __bf16* WT;
  const float* bias;
  if (vh < HH)           { WT = WqT + (size_t)vh * DD * EE;             bias = bq + vh * DD; }
  else if (vh < HH + GG) { WT = WkT + (size_t)(vh - HH) * DD * EE;      bias = bk + (vh - HH) * DD; }
  else                   { WT = WvT + (size_t)(vh - HH - GG) * DD * EE; bias = bv + (vh - HH - GG) * DD; }

  const __bf16* xrow = &xtile[n][0];                // A row (token) in LDS
  const __bf16* wcol = WT + (size_t)(n0 + n) * EE;  // B col (out dim)

  v8f c;
  const float bini = bias[n0 + n];  // column-constant bias across all 8 rows
#pragma unroll
  for (int r = 0; r < 8; ++r) c[r] = bini;

  for (int e0 = 0; e0 < EE; e0 += 32) {
    v16bf a = load_a_bf16(xrow + e0, half);
    v16bf w = load_b_bf16(wcol + e0, half);
    c = wmma_bf16(a, w, c);
  }

  // C layout: VGPR r -> token row s0 + r + half*8, column n0 + n.
  if (vh < HH) {
    __bf16* qp = Qb + (((size_t)b * HH + vh) * SS + s0 + half * 8) * DD + n0 + n;
#pragma unroll
    for (int r = 0; r < 8; ++r) qp[(size_t)r * DD] = (__bf16)c[r];
  } else if (vh < HH + GG) {
    __bf16* kp = Kb + (((size_t)b * GG + (vh - HH)) * SS + s0 + half * 8) * DD + n0 + n;
#pragma unroll
    for (int r = 0; r < 8; ++r) kp[(size_t)r * DD] = (__bf16)c[r];
  } else {  // V stored transposed: Vt[b, g, d, s] -> contiguous 8 tokens per lane
    v8bf pk;
#pragma unroll
    for (int r = 0; r < 8; ++r) pk[r] = (__bf16)c[r];
    __bf16* vp = Vt + (((size_t)b * GG + (vh - HH - GG)) * DD + n0 + n) * SS +
                 s0 + half * 8;
    *(v8bf*)vp = pk;
  }
}

// ---------------- kernel 3: causal flash attention (one wave per q-block) ----
// Scores computed transposed (S^T = K x Q^T) so the P^T C-fragment IS the
// P x V A-fragment: no transpose, no LDS bounce.
// grid: (S/128, H, B), block 256 = 8 waves, wave w -> q0 = (bx*8+w)*16
__global__ void __launch_bounds__(256) k_attn(
    const __bf16* __restrict__ Qb, const __bf16* __restrict__ Kb,
    const __bf16* __restrict__ Vt, float* __restrict__ out) {
  const int lane = threadIdx.x & 31, wave = threadIdx.x >> 5;
  const int n = lane & 15, half = lane >> 4;
  const int h = blockIdx.y, b = blockIdx.z;
  const int g = h / (HH / GG);
  const int q0 = (blockIdx.x * 8 + wave) * 16;
  const int q_abs = q0 + n;  // this lane's q column in S^T

  const __bf16* qrow  = Qb + (((size_t)b * HH + h) * SS + q_abs) * DD;
  const __bf16* kbase = Kb + ((size_t)b * GG + g) * SS * DD;
  const __bf16* vbase = Vt + ((size_t)b * GG + g) * DD * SS;

  // Q^T B-fragments, one per 32-wide slice of D=128; loaded once.
  v16bf qf[4];
#pragma unroll
  for (int t = 0; t < 4; ++t) qf[t] = load_b_bf16(qrow + 32 * t, half);

  v8f oacc[8];  // O tile 16q x 128d: lane = d column, VGPR r = q row r+half*8
#pragma unroll
  for (int t = 0; t < 8; ++t)
#pragma unroll
    for (int r = 0; r < 8; ++r) oacc[t][r] = 0.f;

  float m = NEG, l = 0.f;  // per-lane online-softmax state for q column `n`

  for (int kv0 = 0; kv0 < q0 + 16; kv0 += 32) {
    v8f st0, st1;  // S^T tiles: rows kv0+[0,16)+.., cols = q
#pragma unroll
    for (int r = 0; r < 8; ++r) { st0[r] = 0.f; st1[r] = 0.f; }
#pragma unroll
    for (int t = 0; t < 4; ++t) {
      v16bf a0 = load_a_bf16(kbase + (size_t)(kv0 + n) * DD + 32 * t, half);
      st0 = wmma_bf16(a0, qf[t], st0);
      v16bf a1 = load_a_bf16(kbase + (size_t)(kv0 + 16 + n) * DD + 32 * t, half);
      st1 = wmma_bf16(a1, qf[t], st1);
    }
    // scale + branchless causal mask + partial row max (kv lives in VGPR idx)
    float mloc = NEG;
#pragma unroll
    for (int r = 0; r < 8; ++r) {
      float v0 = st0[r] * SCALE;
      if (kv0 + r + half * 8 > q_abs) v0 = NEG;
      st0[r] = v0; mloc = fmaxf(mloc, v0);
      float v1 = st1[r] * SCALE;
      if (kv0 + 16 + r + half * 8 > q_abs) v1 = NEG;
      st1[r] = v1; mloc = fmaxf(mloc, v1);
    }
    mloc = fmaxf(mloc, __shfl_xor(mloc, 16, 32));  // other half of this q row
    const float mnew = fmaxf(m, mloc);
    const float corr = __expf(m - mnew);
    float rs = 0.f;
#pragma unroll
    for (int r = 0; r < 8; ++r) {
      st0[r] = __expf(st0[r] - mnew); rs += st0[r];
      st1[r] = __expf(st1[r] - mnew); rs += st1[r];
    }
    rs += __shfl_xor(rs, 16, 32);
    l = l * corr + rs;
    m = mnew;

    // P^T C-fragment -> P x V A-fragment: elementwise bf16 pack, no shuffles.
    v16bf pa;
#pragma unroll
    for (int r = 0; r < 8; ++r) { pa[r] = (__bf16)st0[r]; pa[8 + r] = (__bf16)st1[r]; }

    // Rescale O rows: correction for q row j is held by lane j (and j+16).
    float crow[8];
#pragma unroll
    for (int r = 0; r < 8; ++r) crow[r] = __shfl(corr, r + half * 8, 32);
#pragma unroll
    for (int t = 0; t < 8; ++t) {
#pragma unroll
      for (int r = 0; r < 8; ++r) oacc[t][r] *= crow[r];
      v16bf bvf = load_b_bf16(vbase + (size_t)(16 * t + n) * SS + kv0, half);
      oacc[t] = wmma_bf16(pa, bvf, oacc[t]);
    }
  }

  float lrow[8];
#pragma unroll
  for (int r = 0; r < 8; ++r) lrow[r] = __shfl(l, r + half * 8, 32);

#pragma unroll
  for (int t = 0; t < 8; ++t) {
    const int d = 16 * t + n;
#pragma unroll
    for (int r = 0; r < 8; ++r) {
      const int q = q0 + r + half * 8;
      out[((size_t)b * SS + q) * EE + h * DD + d] = oacc[t][r] / lrow[r];
    }
  }
}

// ---------------- launcher ---------------------------------------------------
extern "C" void kernel_launch(void* const* d_in, const int* in_sizes, int n_in,
                              void* d_out, int out_size, void* d_ws, size_t ws_size,
                              hipStream_t stream) {
  (void)in_sizes; (void)n_in; (void)out_size; (void)ws_size;
  const float* x  = (const float*)d_in[0];
  const float* Wq = (const float*)d_in[1];
  const float* bq = (const float*)d_in[2];
  const float* Wk = (const float*)d_in[3];
  const float* bk = (const float*)d_in[4];
  const float* Wv = (const float*)d_in[5];
  const float* bv = (const float*)d_in[6];
  float* out = (float*)d_out;

  char* ws = (char*)d_ws;
  size_t off = 0;
  auto take = [&](size_t bytes) -> char* {
    char* p = ws + off;
    off = (off + bytes + 255) & ~(size_t)255;
    return p;
  };
  __bf16* xbf = (__bf16*)take((size_t)BB * SS * EE * 2);       // [B,S,E]
  __bf16* WqT = (__bf16*)take((size_t)HH * DD * EE * 2);       // [H,D,E]
  __bf16* WkT = (__bf16*)take((size_t)GG * DD * EE * 2);       // [G,D,E]
  __bf16* WvT = (__bf16*)take((size_t)GG * DD * EE * 2);       // [G,D,E]
  __bf16* Qb  = (__bf16*)take((size_t)BB * HH * SS * DD * 2);  // [B,H,S,D]
  __bf16* Kb  = (__bf16*)take((size_t)BB * GG * SS * DD * 2);  // [B,G,S,D]
  __bf16* Vt  = (__bf16*)take((size_t)BB * GG * DD * SS * 2);  // [B,G,D,S]

  k_xconv<<<dim3((unsigned)((size_t)BB * SS * EE / (256 * 8))), 256, 0, stream>>>(
      x, xbf);

  dim3 tb(32, 8);
  k_wtrans<<<dim3(DD / 32, EE / 32, HH), tb, 0, stream>>>(Wq, WqT);
  k_wtrans<<<dim3(DD / 32, EE / 32, GG), tb, 0, stream>>>(Wk, WkT);
  k_wtrans<<<dim3(DD / 32, EE / 32, GG), tb, 0, stream>>>(Wv, WvT);

  k_qkv<<<dim3(SS / 16, HH + 2 * GG, BB), 256, 0, stream>>>(
      xbf, WqT, WkT, WvT, bq, bk, bv, Qb, Kb, Vt);

  k_attn<<<dim3(SS / 128, HH, BB), 256, 0, stream>>>(Qb, Kb, Vt, out);
}
